// GATLabelEncoder_1846835937698
// MI455X (gfx1250) — matrix operand order repaired and checked by reading, not compiled, not run on hardware
//
#include <hip/hip_runtime.h>
#include <hip/hip_bf16.h>
#include <math.h>

#define HID   256
#define D_INF 160

typedef __attribute__((ext_vector_type(16))) __bf16 v16bf;
typedef __attribute__((ext_vector_type(8)))  float  v8f;

__device__ __forceinline__ unsigned f2bf_rne(float f) {
    unsigned u = __float_as_uint(f);
    unsigned r = u + 0x7FFFu + ((u >> 16) & 1u);
    return r >> 16;
}

// monotone float<->uint key for atomic max over signed floats
__device__ __forceinline__ unsigned fkey(float f) {
    unsigned u = __float_as_uint(f);
    return u ^ ((unsigned)((int)u >> 31) | 0x80000000u);
}
__device__ __forceinline__ float funkey(unsigned u) {
    unsigned m = (u & 0x80000000u) ? 0x80000000u : 0xFFFFFFFFu;
    return __uint_as_float(u ^ m);
}

// ---------------------------------------------------------------- utilities
__global__ void fill_u32_kernel(unsigned* p, unsigned v, long long n) {
    long long i = (long long)blockIdx.x * blockDim.x + threadIdx.x;
    if (i < n) p[i] = v;
}

// ---------------------------------------------------------------- embedding gather -> bf16 X
__global__ void embed_kernel(const int* __restrict__ ids, const int* __restrict__ lev,
                             const float* __restrict__ nemb, const float* __restrict__ lemb,
                             unsigned short* __restrict__ Xbf, int N) {
    long long t = (long long)blockIdx.x * blockDim.x + threadIdx.x;
    if (t >= (long long)N * D_INF) return;
    int n = (int)(t / D_INF), c = (int)(t % D_INF);
    float v = (c < 128) ? nemb[(long long)ids[n] * 128 + c]
                        : lemb[(long long)lev[n] * 32 + (c - 128)];
    Xbf[(long long)n * HID + c] = (unsigned short)f2bf_rne(v);
}

// ---------------------------------------------------------------- pack W[K,256] f32 -> per-lane WMMA-B fragments
// Record layout: Bswz[((kk/32)*16 + tn)*32 + lane] = 8 consecutive u32 (= v16bf fragment for that lane).
// B 32x16 bf16 fragment, contiguous K (same order as A): lane -> col N = tn*16 + (lane&15), khalf = lane>>4;
// u32 i holds packed K = kk + khalf*16 + 2i, +1 for that column.
__global__ void packW_swz_kernel(const float* __restrict__ W, unsigned* __restrict__ Bswz, int Kdim) {
    int t = blockIdx.x * blockDim.x + threadIdx.x;          // one u32 of output per thread
    int total = (Kdim / 2) * HID;                           // == (Kdim/32)*16*32*8
    if (t >= total) return;
    int i     = t & 7;
    int lane  = (t >> 3) & 31;
    int tn    = (t >> 8) & 15;
    int kblk  = t >> 12;
    int khalf = lane >> 4;
    int col   = tn * 16 + (lane & 15);
    int k0    = kblk * 32 + khalf * 16 + 2 * i;
    unsigned lo = f2bf_rne(W[(long long)k0       * HID + col]);
    unsigned hi = f2bf_rne(W[(long long)(k0 + 1) * HID + col]);
    Bswz[t] = lo | (hi << 16);
}

// ---------------------------------------------------------------- WMMA GEMM: H = X(bf16) @ W(bf16), f32 accum
// one wave per 16x16 output tile; K-loop of v_wmma_f32_16x16x32_bf16
// A: 2x b128 per lane from row-major X; B: 2x b128 per lane from pre-swizzled fragments.
__global__ void __launch_bounds__(256)
gemm_wmma_kernel(const unsigned short* __restrict__ Xbf, const unsigned* __restrict__ Bswz,
                 float* __restrict__ Hout, int tiles_m, int Kdim) {
    int wave = (int)(((long long)blockIdx.x * blockDim.x + threadIdx.x) >> 5);
    int lane = threadIdx.x & 31;
    const int tiles_n = HID / 16;
    int tm = wave / tiles_n, tn = wave % tiles_n;
    if (tm >= tiles_m) return;            // whole-wave uniform exit; EXEC stays all-ones

    int r = lane & 15;
    int khalf = lane >> 4;                // 0 or 1
    const uint4* Xv = (const uint4*)((const unsigned*)Xbf
                       + (long long)(tm * 16 + r) * (HID / 2) + khalf * 8);
    const uint4* Bv = (const uint4*)Bswz + ((long long)tn * 32 + lane) * 2;

    v8f c = {};
    for (int kk = 0; kk < Kdim; kk += 32) {
        union { uint4 q[2]; v16bf v; } a, b;
        // A 16x32 bf16 layout: one contiguous 32B group (K=khalf*16..khalf*16+15)
        a.q[0] = Xv[(kk >> 4) + 0];       // pairs kk/4 + khalf*8 .. +3
        a.q[1] = Xv[(kk >> 4) + 1];       // pairs kk/4 + khalf*8 + 4 .. +7
        // B fragment: one 32B record per lane, coalesced across the wave
        const uint4* br = Bv + (long long)(kk >> 5) * (16 * 32 * 2);
        b.q[0] = br[0];
        b.q[1] = br[1];

        c = __builtin_amdgcn_wmma_f32_16x16x32_bf16(
                /*neg_a=*/false, a.v, /*neg_b=*/false, b.v,
                /*c_mod=*/(short)0, c, /*reuse_a=*/false, /*reuse_b=*/false);
    }
    // C/D f32 16x16 layout: VGPR v -> row M = khalf*8 + v, col N = lane&15
    int col = tn * 16 + r;
    #pragma unroll
    for (int i = 0; i < 8; ++i)
        Hout[(long long)(tm * 16 + khalf * 8 + i) * HID + col] = c[i];
}

// ---------------------------------------------------------------- per-(node,head) attention coefficients
__global__ void attn_coef_kernel(const float* __restrict__ Hm,
                                 const float* __restrict__ as_w, const float* __restrict__ ad_w,
                                 float* __restrict__ a_s, float* __restrict__ a_d,
                                 int N, int heads, int ch) {
    int t = blockIdx.x * blockDim.x + threadIdx.x;
    if (t >= N * heads) return;
    int n = t / heads, hh = t % heads;
    const float* hp = Hm + (long long)n * HID + hh * ch;
    float s1 = 0.f, s2 = 0.f;
    for (int cc = 0; cc < ch; ++cc) {
        float hv = hp[cc];
        s1 += hv * as_w[hh * ch + cc];
        s2 += hv * ad_w[hh * ch + cc];
    }
    a_s[t] = s1; a_d[t] = s2;
}

// ---------------------------------------------------------------- edge pass 1: leaky-relu + segment max
__global__ void edge_max_kernel(const int* __restrict__ ei, int E, int heads,
                                const float* __restrict__ a_s, const float* __restrict__ a_d,
                                float* __restrict__ exbuf, unsigned* __restrict__ mkey,
                                long long total) {
    long long t = (long long)blockIdx.x * blockDim.x + threadIdx.x;
    if (t >= total) return;
    long long e = t / heads;
    int hh = (int)(t % heads);
    int src = (e < E) ? ei[e] : (int)(e - E);
    int dst = (e < E) ? ei[(long long)E + e] : (int)(e - E);
    float v = a_s[(long long)src * heads + hh] + a_d[(long long)dst * heads + hh];
    v = (v > 0.f) ? v : 0.2f * v;                 // leaky_relu(0.2)
    exbuf[t] = v;
    atomicMax(mkey + (long long)dst * heads + hh, fkey(v));
}

// ---------------------------------------------------------------- edge pass 2: exp + segment sum
__global__ void edge_exp_kernel(const int* __restrict__ ei, int E, int heads,
                                const unsigned* __restrict__ mkey,
                                float* __restrict__ exbuf, float* __restrict__ den,
                                long long total) {
    long long t = (long long)blockIdx.x * blockDim.x + threadIdx.x;
    if (t >= total) return;
    long long e = t / heads;
    int hh = (int)(t % heads);
    int dst = (e < E) ? ei[(long long)E + e] : (int)(e - E);
    float m = funkey(mkey[(long long)dst * heads + hh]);
    float x = __expf(exbuf[t] - m);
    exbuf[t] = x;
    atomicAdd(den + (long long)dst * heads + hh, x);
}

// ---------------------------------------------------------------- edge pass 3: weighted message aggregation
__global__ void edge_aggr_kernel(const int* __restrict__ ei, int E, int heads, int ch,
                                 const float* __restrict__ exbuf, const float* __restrict__ den,
                                 const float* __restrict__ Hm, float* __restrict__ acc,
                                 long long total) {
    long long t = (long long)blockIdx.x * blockDim.x + threadIdx.x;
    if (t >= total) return;
    long long e = t >> 6;
    int q = (int)(t & 63);
    int c0 = q * 4;
    int src = (e < E) ? ei[e] : (int)(e - E);
    int dst = (e < E) ? ei[(long long)E + e] : (int)(e - E);
    int hh = c0 / ch;
    float alpha = exbuf[e * heads + hh] / den[(long long)dst * heads + hh];
    const float4 hv = *(const float4*)(Hm + (long long)src * HID + c0);
    float* ap = acc + (long long)dst * HID + c0;
    atomicAdd(ap + 0, hv.x * alpha);
    atomicAdd(ap + 1, hv.y * alpha);
    atomicAdd(ap + 2, hv.z * alpha);
    atomicAdd(ap + 3, hv.w * alpha);
}

// ---------------------------------------------------------------- +bias, ELU, write f32 + bf16 for next layer
__global__ void epilogue_kernel(const float* __restrict__ acc, const float* __restrict__ bias,
                                float* __restrict__ Xf, unsigned short* __restrict__ Xbf,
                                long long total) {
    long long t = (long long)blockIdx.x * blockDim.x + threadIdx.x;
    if (t >= total) return;
    int c = (int)(t % HID);
    float v = acc[t] + bias[c];
    v = (v > 0.f) ? v : (__expf(v) - 1.f);        // ELU
    Xf[t] = v;
    Xbf[t] = (unsigned short)f2bf_rne(v);
}

// ---------------------------------------------------------------- final: lw * x, proj, LayerNorm on rows 0..7
__global__ void __launch_bounds__(256)
final_kernel(const float* __restrict__ Xf, const int* __restrict__ levels,
             const float* __restrict__ lvw, const float* __restrict__ projW,
             const float* __restrict__ projb, const float* __restrict__ ln_g,
             const float* __restrict__ ln_b, float* __restrict__ out, int out_size) {
    __shared__ float y[8 * HID];
    __shared__ float rs[HID], rq[HID];
    int c = threadIdx.x;

    // softmax over 9 level weights (redundant per thread; trivial)
    float lw[9];
    float mx = lvw[0];
    for (int i = 1; i < 9; ++i) mx = fmaxf(mx, lvw[i]);
    float s = 0.f;
    for (int i = 0; i < 9; ++i) { lw[i] = __expf(lvw[i] - mx); s += lw[i]; }
    float inv = 1.f / s;
    for (int i = 0; i < 9; ++i) lw[i] *= inv;

    for (int n = 0; n < 8; ++n)
        y[n * HID + c] = Xf[(long long)n * HID + c] * lw[levels[n]];
    __syncthreads();

    for (int n = 0; n < 8; ++n) {
        float a = projb[c];
        for (int k = 0; k < HID; ++k)
            a += y[n * HID + k] * projW[(long long)k * HID + c];
        rs[c] = a; rq[c] = a * a;
        __syncthreads();
        for (int off = 128; off > 0; off >>= 1) {
            if (c < off) { rs[c] += rs[c + off]; rq[c] += rq[c + off]; }
            __syncthreads();
        }
        float mu  = rs[0] * (1.f / HID);
        float var = rq[0] * (1.f / HID) - mu * mu;
        float rsd = rsqrtf(var + 1e-5f);
        if (n * HID + c < out_size)
            out[n * HID + c] = (a - mu) * rsd * ln_g[c] + ln_b[c];
        __syncthreads();
    }
    // mask (all false) + any trailing outputs -> zero
    if (8 * HID + c < out_size) out[8 * HID + c] = 0.f;
}

// ================================================================ host launcher
static inline int ceil_div_ll(long long a, int b) { return (int)((a + b - 1) / b); }

extern "C" void kernel_launch(void* const* d_in, const int* in_sizes, int n_in,
                              void* d_out, int out_size, void* d_ws, size_t ws_size,
                              hipStream_t stream) {
    const int*   node_ids    = (const int*)d_in[0];
    const int*   node_levels = (const int*)d_in[1];
    const int*   ei          = (const int*)d_in[2];
    const float* node_emb    = (const float*)d_in[3];
    const float* level_emb   = (const float*)d_in[4];
    const float* level_w     = (const float*)d_in[5];
    const float* W[3]  = { (const float*)d_in[6],  (const float*)d_in[10], (const float*)d_in[14] };
    const float* As[3] = { (const float*)d_in[7],  (const float*)d_in[11], (const float*)d_in[15] };
    const float* Ad[3] = { (const float*)d_in[8],  (const float*)d_in[12], (const float*)d_in[16] };
    const float* Bb[3] = { (const float*)d_in[9],  (const float*)d_in[13], (const float*)d_in[17] };
    const float* projW = (const float*)d_in[18];
    const float* projb = (const float*)d_in[19];
    const float* ln_g  = (const float*)d_in[20];
    const float* ln_b  = (const float*)d_in[21];
    float* out = (float*)d_out;

    const int N      = in_sizes[0];
    const int E      = in_sizes[2] / 2;
    const long long EDGES = (long long)E + N;
    const int N_pad  = (N + 15) & ~15;
    const int BLK    = 256;

    // workspace carve-out
    char* w = (char*)d_ws;
    auto carve = [&](size_t bytes) { char* p = w; w += (bytes + 255) & ~(size_t)255; return p; };
    unsigned short* Xbf  = (unsigned short*)carve((size_t)N_pad * HID * 2);
    float*          Xf   = (float*)carve((size_t)N_pad * HID * 4);
    float*          Hm   = (float*)carve((size_t)N_pad * HID * 4);
    float*          acc  = (float*)carve((size_t)N_pad * HID * 4);
    unsigned*       Bswz = (unsigned*)carve((size_t)(HID / 2) * HID * 4);
    float*          a_s  = (float*)carve((size_t)N * 4 * 4);
    float*          a_d  = (float*)carve((size_t)N * 4 * 4);
    unsigned*       mkey = (unsigned*)carve((size_t)N * 4 * 4);
    float*          den  = (float*)carve((size_t)N * 4 * 4);
    float*          exbf = (float*)carve((size_t)EDGES * 4 * 4);
    (void)ws_size; (void)n_in;

    // zero bf16 X (covers zero-padded K cols of layer 0 and padded rows)
    {
        long long cnt = (long long)N_pad * (HID / 2);
        fill_u32_kernel<<<ceil_div_ll(cnt, BLK), BLK, 0, stream>>>((unsigned*)Xbf, 0u, cnt);
    }
    // embedding gather
    {
        long long cnt = (long long)N * D_INF;
        embed_kernel<<<ceil_div_ll(cnt, BLK), BLK, 0, stream>>>(node_ids, node_levels,
                                                                node_emb, level_emb, Xbf, N);
    }

    const int tiles_m = N_pad / 16;
    for (int l = 0; l < 3; ++l) {
        const int Kdim  = (l == 0) ? D_INF : HID;
        const int heads = (l == 0) ? 4 : 1;
        const int ch    = (l == 0) ? 64 : HID;

        // pack weights into per-lane WMMA-B fragment records (coalesced b128 loads in GEMM)
        {
            int cnt = (Kdim / 2) * HID;
            packW_swz_kernel<<<ceil_div_ll(cnt, BLK), BLK, 0, stream>>>(W[l], Bswz, Kdim);
        }
        // H = X @ W via WMMA (one wave per 16x16 tile)
        {
            long long thr = (long long)tiles_m * 16 * 32;
            gemm_wmma_kernel<<<ceil_div_ll(thr, BLK), BLK, 0, stream>>>(Xbf, Bswz, Hm, tiles_m, Kdim);
        }
        // attention coefficients per (node, head)
        {
            int cnt = N * heads;
            attn_coef_kernel<<<ceil_div_ll(cnt, BLK), BLK, 0, stream>>>(Hm, As[l], Ad[l],
                                                                        a_s, a_d, N, heads, ch);
        }
        // init segment max keys (-inf) and denominators (0)
        {
            long long cnt = (long long)N * heads;
            fill_u32_kernel<<<ceil_div_ll(cnt, BLK), BLK, 0, stream>>>(mkey, 0u, cnt);
            fill_u32_kernel<<<ceil_div_ll(cnt, BLK), BLK, 0, stream>>>((unsigned*)den, 0u, cnt);
        }
        // edge passes
        {
            long long cnt = EDGES * heads;
            edge_max_kernel<<<ceil_div_ll(cnt, BLK), BLK, 0, stream>>>(ei, E, heads, a_s, a_d,
                                                                       exbf, mkey, cnt);
            edge_exp_kernel<<<ceil_div_ll(cnt, BLK), BLK, 0, stream>>>(ei, E, heads, mkey,
                                                                       exbf, den, cnt);
        }
        // zero accumulator, aggregate messages
        {
            long long cnt = (long long)N * HID;
            fill_u32_kernel<<<ceil_div_ll(cnt, BLK), BLK, 0, stream>>>((unsigned*)acc, 0u, cnt);
            long long agg = EDGES * 64;
            edge_aggr_kernel<<<ceil_div_ll(agg, BLK), BLK, 0, stream>>>(ei, E, heads, ch,
                                                                        exbf, den, Hm, acc, agg);
            epilogue_kernel<<<ceil_div_ll(cnt, BLK), BLK, 0, stream>>>(acc, Bb[l], Xf, Xbf, cnt);
        }
    }

    // level weighting + projection + LayerNorm on the 8 output rows, plus zero mask
    final_kernel<<<1, BLK, 0, stream>>>(Xf, node_levels, level_w, projW, projb,
                                        ln_g, ln_b, out, out_size);
}
